// NetBoW_23106924053264
// MI455X (gfx1250) — matrix operand order repaired and checked by reading, not compiled, not run on hardware
//
#include <hip/hip_runtime.h>
#include <hip/hip_bf16.h>
#include <math.h>

// ---------------------------------------------------------------------------
// NetBoW on MI455X (gfx1250, wave32)
//   x:        (8, 128, 64, 64) f32
//   centroids:(2048, 128)      f32
//   out:      (8, 2048)        f32  (L2-normalized bag-of-words rows)
//
// Compute-bound VALU kernel (|a-b|+acc, 2 ops/elem); all data L2-resident.
// CDNA5 paths: TENSOR_LOAD_TO_LDS (TDM, double-buffered, LDS row padding via
// descriptor pad fields), s_wait_tensorcnt, ds_add_f32 LDS atomics, and a
// v_wmma_f32_16x16x32_f16 Gram-diagonal for the final row norms.
//
// Softmax over K=2048 uses a two-pass online scheme per location (running
// min + rescaled sum, then a second streaming pass to accumulate), so all
// loops stay rolled: small code, low register pressure, one TDM call site
// per pipeline stage.
// ---------------------------------------------------------------------------

#define NIMG   8
#define C      128
#define HW     4096          // 64*64
#define K      2048
#define KT     32            // centroids per tile (1 per lane)
#define NT     (K / KT)      // 64 tiles
#define CEN_ROW 132          // 128 dwords + 4 pad dwords: bank spread, 16B aligned
#define ROUNDS 4             // location-rounds per block (8 locations each)
#define BETA   (1000.0f / 128.0f)

// LDS layout (floats) inside one static __shared__ array (assumed LDS offset 0
// so the TDM descriptor's lds_addr matches C++-side accesses).
#define CEN0_F   0
#define CEN1_F   (KT * CEN_ROW)                 // 4224
#define XOFF_F   (2 * KT * CEN_ROW)            // 8448
#define BOG_F    (XOFF_F + 8 * C)              // 9472
#define SMEM_F   (BOG_F + K)                   // 11520 floats = 46.1 KB

typedef unsigned int u32x4 __attribute__((ext_vector_type(4)));
typedef int          i32x8 __attribute__((ext_vector_type(8)));
typedef int          i32x4 __attribute__((ext_vector_type(4)));
typedef float        v8f   __attribute__((ext_vector_type(8)));
typedef _Float16     v16h  __attribute__((ext_vector_type(16)));

// One TDM load: 2D tile (128 f32 x KT rows) global->LDS, inserting 4 pad
// dwords every 128 dwords (row stride CEN_ROW=132). D# per CDNA5 ISA §8.3/8.4.
// Whole-wave instruction; call from wave 0 only. 6-arg builtin on this
// toolchain: (u32x4 g0, i32x8 g1, i32x4 g2, i32x4 g3, i32x8 g4, i32 cpol).
__device__ __forceinline__ void tdm_load_tile(unsigned long long gaddr,
                                              unsigned int lds_byte_off) {
  u32x4 g0;
  g0.x = 0x1u;                                   // count=1, user descriptor
  g0.y = lds_byte_off;                           // lds_addr (bytes)
  g0.z = (unsigned int)(gaddr & 0xFFFFFFFFull);  // global_addr[31:0]
  g0.w = (unsigned int)((gaddr >> 32) & 0x01FFFFFFull) // global_addr[56:32]
       | 0x80000000u;                            // type=2 in bits [127:126]

  i32x8 g1;
  // data_size=2 (4B)<<16 | pad_enable<<20 | pad_interval=6 (128 dw)<<22
  // | pad_amount=3 (4 dw)<<25
  g1[0] = (int)0x07920000u;
  g1[1] = (int)(128u << 16);          // tensor_dim0 = 128
  g1[2] = (int)(2048u << 16);         // tensor_dim1 = 2048
  g1[3] = (int)(128u << 16);          // tile_dim0 = 128
  g1[4] = (int)(unsigned)KT;          // tile_dim1 = 32, tile_dim2 = 0
  g1[5] = 128;                        // tensor_dim0_stride = 128
  g1[6] = 0;
  g1[7] = 0;

  i32x4 gz4 = {0, 0, 0, 0};           // 2D tensor: groups 2/3 unused
  i32x8 gz8 = {0, 0, 0, 0, 0, 0, 0, 0};
  __builtin_amdgcn_tensor_load_to_lds(g0, g1, gz4, gz4, gz8, 0);
}

// L1 distance (sum over 128 dims) between x row (LDS, broadcast reads) and
// this lane's centroid row in the given LDS tile buffer.
__device__ __forceinline__ float l1_dist(const float4* __restrict__ xr4,
                                         const float* __restrict__ cenbuf,
                                         int lane) {
  const float4* cr4 =
      reinterpret_cast<const float4*>(cenbuf + lane * CEN_ROW);
  float a0 = 0.f, a1 = 0.f, a2 = 0.f, a3 = 0.f;
  for (int c = 0; c < C / 4; ++c) {   // ds_load_b128 x2 + 8 VALU per iter
    float4 xv = xr4[c];
    float4 cv = cr4[c];
    a0 += fabsf(xv.x - cv.x);
    a1 += fabsf(xv.y - cv.y);
    a2 += fabsf(xv.z - cv.z);
    a3 += fabsf(xv.w - cv.w);
  }
  return (a0 + a1) + (a2 + a3);
}

__global__ void __launch_bounds__(256)
netbow_main(const float* __restrict__ x, const float* __restrict__ cen,
            float* __restrict__ bog_ws) {
  __shared__ float smem[SMEM_F];

  const int tid  = threadIdx.x;
  const int lane = tid & 31;
  const int wid  = tid >> 5;            // 8 waves: one location per wave/round
  const int bi   = blockIdx.x & 127;    // block within image
  const int n    = blockIdx.x >> 7;     // image index (8 * 128 = 1024 blocks)

  for (int i = tid; i < K; i += 256) smem[BOG_F + i] = 0.0f;  // LDS bog accum

  const unsigned long long cen_base = (unsigned long long)(uintptr_t)cen;

  for (int r = 0; r < ROUNDS; ++r) {
    // ---- load + L2-normalize this wave's descriptor into LDS -------------
    const int hw = bi * 32 + r * 8 + wid;
    const float* xp = x + (size_t)n * C * HW + hw;
    float v0 = xp[(size_t)(lane +  0) * HW];
    float v1 = xp[(size_t)(lane + 32) * HW];
    float v2 = xp[(size_t)(lane + 64) * HW];
    float v3 = xp[(size_t)(lane + 96) * HW];
    float ss = v0 * v0 + v1 * v1 + v2 * v2 + v3 * v3;
    #pragma unroll
    for (int off = 16; off >= 1; off >>= 1) ss += __shfl_xor(ss, off, 32);
    const float inorm = 1.0f / fmaxf(sqrtf(ss), 1e-12f);
    float* xrow = &smem[XOFF_F + wid * C];
    xrow[lane +  0] = v0 * inorm;
    xrow[lane + 32] = v1 * inorm;
    xrow[lane + 64] = v2 * inorm;
    xrow[lane + 96] = v3 * inorm;
    __syncthreads();
    const float4* xr4 = reinterpret_cast<const float4*>(xrow);

    float m_w, invs;
    // ==== two streaming passes over the 64 centroid tiles ==================
    for (int pass = 0; pass < 2; ++pass) {
      // prologue: TDM tile 0 into buffer 0
      if (wid == 0) {
        tdm_load_tile(cen_base, (unsigned int)(CEN0_F * 4));
        __builtin_amdgcn_s_wait_tensorcnt(0);
      }
      __syncthreads();

      float m = 3.0e38f;   // running min of this lane's L1 sums (pass 0)
      float s = 0.0f;      // running sum of exp(-BETA*(res - m))

      for (int t = 0; t < NT; ++t) {          // rolled, double-buffered
        if (wid == 0 && t + 1 < NT) {
          tdm_load_tile(cen_base + (unsigned long long)(t + 1) * KT * C * 4,
                        (unsigned int)((((t + 1) & 1) ? CEN1_F : CEN0_F) * 4));
        }
        const float* cbuf = &smem[(t & 1) ? CEN1_F : CEN0_F];
        const float res = l1_dist(xr4, cbuf, lane);

        if (pass == 0) {
          // online min/sum update (reference point = current lane min)
          const float nm = fminf(m, res);
          s = s * __expf(-BETA * (m - nm)) + __expf(-BETA * (res - nm));
          m = nm;
        } else {
          // exact softmax term, accumulate into shared bog (k = t*32 + lane)
          const float p = __expf(-BETA * (res - m_w)) * invs;
          atomicAdd(&smem[BOG_F + t * KT + lane], p);   // ds_add_f32
        }
        __syncthreads();                    // all waves done with buffer t&1
        if (wid == 0) __builtin_amdgcn_s_wait_tensorcnt(0);
        __syncthreads();                    // tile t+1 visible in LDS
      }

      if (pass == 0) {
        // merge per-lane online states across the wave (location-wide softmax)
        float mm = m;
        #pragma unroll
        for (int off = 16; off >= 1; off >>= 1)
          mm = fminf(mm, __shfl_xor(mm, off, 32));
        float sadj = s * __expf(-BETA * (m - mm));  // rescale to global min
        #pragma unroll
        for (int off = 16; off >= 1; off >>= 1)
          sadj += __shfl_xor(sadj, off, 32);
        m_w  = mm;
        invs = 1.0f / sadj;                 // sadj >= 1 always
      }
    }
    __syncthreads();
  }

  // ---- flush LDS bog row into global accumulator -------------------------
  for (int i = tid; i < K; i += 256)
    atomicAdd(&bog_ws[(size_t)n * K + i], smem[BOG_F + i]);
}

// ---------------------------------------------------------------------------
// Row L2 norms of the 8x2048 bog via a WMMA Gram diagonal (f16 inputs, f32
// accumulate; bog entries are softmax sums so f16 relative error ~5e-4 is
// fine for a norm), then normalize into d_out.
//
// v_wmma_f32_16x16x32_f16 operand packing (ISA §7.12.2, 16-bit matrices):
//  A (16x32, M x K): lanes 0-15 hold M=lane with K={0..7,16..23} in elems
//    0..15; lanes 16-31 hold M=lane-16 with K={8..15,24..31}.
//  B (32x16, K x N): lanes 0-15 hold N=lane with K={0..7,16..23}; lanes 16-31 hold
//    N=lane-16 with K={8..15,24..31}.
//  D (16x16 f32): elem j = row M=j (lanes 0-15, N=lane) / M=j+8 (lanes 16-31).
//
// Gram: B[k][n] = bog[n][k], so both operands come from row (lane&15). Since
// the diagonal entries (m,m) for m<8 depend only on A-row m and B-column m,
// rows 8..15 of the padded operand may be *duplicates* of rows 0..7 instead
// of zeros -> row = lane&7, all loads unconditional and contiguous (the
// lane-half K-offset patterns fold into base-pointer selects), so the
// compiler emits clean wide loads instead of per-element divergent branches.
// ---------------------------------------------------------------------------
__global__ void __launch_bounds__(256)
netbow_finalize(const float* __restrict__ bog, float* __restrict__ out) {
  __shared__ float norms[16];
  const int tid  = threadIdx.x;
  const int lane = tid & 31;

  if ((tid >> 5) == 0) {
    const int row = lane & 7;           // rows 8..15 duplicate rows 0..7
    const int hi  = (lane >= 16) ? 1 : 0;
    v8f acc = {};
    for (int kb = 0; kb < K; kb += 32) {
      const float* rp  = bog + (size_t)row * K + kb;
      const float* rpA = rp + (hi ? 8 : 0);    // A: runs {0..7} and {16..23}
      const float* rpB = rp + (hi ? 8 : 0);    // B: runs {0..7} and {16..23}
      v16h a, b;
      #pragma unroll
      for (int j = 0; j < 8; ++j) {
        a[j]     = (_Float16)rpA[j];
        a[j + 8] = (_Float16)rpA[j + 16];
        b[j]     = (_Float16)rpB[j];
        b[j + 8] = (_Float16)rpB[j + 16];
      }
      acc = __builtin_amdgcn_wmma_f32_16x16x32_f16(
          /*neg_a=*/false, a, /*neg_b=*/false, b,
          /*c_mod=*/(short)0, acc, /*reuse_a=*/false, /*reuse_b=*/false);
    }
    // diagonal (m,m), m<8: acc elem m on lane m (lanes 0-15 half)
    float nv = 0.0f;
    #pragma unroll
    for (int j = 0; j < 8; ++j)
      if (lane == j) nv = acc[j];
    if (lane < 8) norms[lane] = nv;
  }
  __syncthreads();

  for (int i = tid; i < NIMG * K; i += 256) {
    const int row = i >> 11;
    out[i] = bog[i] / fmaxf(sqrtf(norms[row]), 1e-12f);
  }
}

__global__ void zero_f32(float* p, int n) {
  int i = blockIdx.x * blockDim.x + threadIdx.x;
  if (i < n) p[i] = 0.0f;
}

extern "C" void kernel_launch(void* const* d_in, const int* in_sizes, int n_in,
                              void* d_out, int out_size, void* d_ws, size_t ws_size,
                              hipStream_t stream) {
  (void)in_sizes; (void)n_in; (void)out_size; (void)ws_size;
  const float* x   = (const float*)d_in[0];   // (8,128,64,64) f32
  const float* cen = (const float*)d_in[1];   // (2048,128)    f32
  float* out    = (float*)d_out;              // (8,2048)      f32
  float* bog_ws = (float*)d_ws;               // 64 KB scratch accumulator

  zero_f32<<<dim3((NIMG * K + 255) / 256), dim3(256), 0, stream>>>(bog_ws, NIMG * K);
  netbow_main<<<dim3(NIMG * 128), dim3(256), 0, stream>>>(x, cen, bog_ws);
  netbow_finalize<<<dim3(1), dim3(256), 0, stream>>>(bog_ws, out);
}
